// MoELayer_33921651704704
// MI455X (gfx1250) — compile-verified, matrix operand order checked by reading, not run on hardware
//
#include <hip/hip_runtime.h>
#include <hip/hip_bf16.h>

// Problem constants
#define TOK  4096      // B*S
#define DIM  1024      // D
#define HID  2048      // H
#define OUTD 1024      // O
#define NEXP 8
#define CAP  10240     // 2*TOK slots + per-expert 256-alignment padding (40*256)
#define EPSV 1e-5f
#define INV9 (1.0f / 9.0f)

// GEMM tile: 256 threads = 8 wave32; block tile 256x64, K-step 32.
// Each wave owns a 32x64 strip of C: 2 A-frags x 4 B-frags = 8 WMMAs/K-step.
#define BM 256
#define BN 64
#define BK 32
#define LDA 40   // halves per LDS row (pad keeps 16B alignment, spreads banks)
#define LDB 40

typedef _Float16 v16h __attribute__((ext_vector_type(16)));
typedef float    v8f  __attribute__((ext_vector_type(8)));

union FragU { v16h h; int4 q[2]; };

// 16-bit A/B fragment: lane-half selects K-group of 8; halves [0..7] cover
// K = kb..kb+7, halves [8..15] cover K = 16+kb..16+kb+7
// -> two 16-byte ds_load_b128 per fragment.
__device__ inline v16h ld_frag(const _Float16* row, int kb) {
  FragU f;
  f.q[0] = *(const int4*)(row + kb);
  f.q[1] = *(const int4*)(row + 16 + kb);
  return f.h;
}

// CDNA5 async global->LDS copy (16B per lane), tracked by ASYNCcnt.
__device__ inline void async_copy16(unsigned lds_off, const void* gptr) {
  asm volatile("global_load_async_to_lds_b128 %0, %1, off"
               :: "v"(lds_off), "v"((unsigned long long)(uintptr_t)gptr)
               : "memory");
}
#define WAIT_ASYNC() asm volatile("s_wait_asynccnt 0" ::: "memory")

__device__ inline float wave_sum(float v) {
  #pragma unroll
  for (int o = 16; o > 0; o >>= 1) v += __shfl_down(v, o, 32);
  return v;
}

// ---------------------------------------------------------------- init
__global__ void init_kernel(int* __restrict__ tok_id, int* __restrict__ meta) {
  int i = blockIdx.x * 256 + threadIdx.x;
  if (i < CAP) tok_id[i] = -1;
  if (i < 32) meta[i] = 0;   // counts[0:8), cursor[8:16), seg[16:25)
}

// ------------------------------------------------- LN1 + router + top-2
__global__ __launch_bounds__(256)
void ln_router_kernel(const float* __restrict__ x,
                      const float* __restrict__ g1, const float* __restrict__ b1,
                      const float* __restrict__ rw, const float* __restrict__ rb,
                      _Float16* __restrict__ xh,
                      int* __restrict__ tk_i, float* __restrict__ tk_w,
                      int* __restrict__ counts) {
  __shared__ float xs[DIM];
  __shared__ float red[16];
  __shared__ float logits[NEXP];
  const int t = blockIdx.x, tid = threadIdx.x;
  const int wid = tid >> 5, lane = tid & 31;
  const float* xr = x + (size_t)t * DIM;

  float xl[4], sum = 0.f, sq = 0.f;
  #pragma unroll
  for (int i = 0; i < 4; i++) {
    float v = xr[tid + i * 256];
    xl[i] = v; sum += v; sq += v * v;
  }
  sum = wave_sum(sum); sq = wave_sum(sq);
  if (lane == 0) { red[wid] = sum; red[8 + wid] = sq; }
  __syncthreads();
  if (tid == 0) {
    float s = 0.f, q = 0.f;
    for (int i = 0; i < 8; i++) { s += red[i]; q += red[8 + i]; }
    red[0] = s; red[8] = q;
  }
  __syncthreads();
  const float mean = red[0] / DIM;
  const float rstd = rsqrtf(red[8] / DIM - mean * mean + EPSV);
  #pragma unroll
  for (int i = 0; i < 4; i++) {
    int d = tid + i * 256;
    float xn = (xl[i] - mean) * rstd * g1[d] + b1[d];
    xs[d] = xn;
    xh[(size_t)t * DIM + d] = (_Float16)xn;
  }
  __syncthreads();

  // one wave per expert logit (exactly 8 waves)
  float p = 0.f;
  const float* wr = rw + (size_t)wid * DIM;
  for (int d = lane; d < DIM; d += 32) p += xs[d] * wr[d];
  p = wave_sum(p);
  if (lane == 0) logits[wid] = p + rb[wid];
  __syncthreads();

  if (tid == 0) {
    float mx = logits[0];
    for (int e = 1; e < NEXP; e++) mx = fmaxf(mx, logits[e]);
    float pe[NEXP], s = 0.f;
    for (int e = 0; e < NEXP; e++) { pe[e] = __expf(logits[e] - mx); s += pe[e]; }
    for (int e = 0; e < NEXP; e++) pe[e] /= s;
    int i0 = 0;
    for (int e = 1; e < NEXP; e++) if (pe[e] > pe[i0]) i0 = e;
    int i1 = (i0 == 0) ? 1 : 0;
    for (int e = 0; e < NEXP; e++) if (e != i0 && pe[e] > pe[i1]) i1 = e;
    float w0 = pe[i0], w1 = pe[i1], ws = w0 + w1;
    tk_i[2 * t] = i0;  tk_i[2 * t + 1] = i1;
    tk_w[2 * t] = w0 / ws;  tk_w[2 * t + 1] = w1 / ws;
    atomicAdd(&counts[i0], 1);
    atomicAdd(&counts[i1], 1);
  }
}

// ------------------------------------------- 256-aligned segment offsets
__global__ void offsets_kernel(int* __restrict__ meta) {
  if (threadIdx.x == 0) {
    int acc = 0;
    for (int e = 0; e < NEXP; e++) {
      meta[16 + e] = acc;
      acc += ((meta[e] + BM - 1) / BM) * BM;
    }
    meta[24] = acc;  // total aligned rows
  }
}

// ------------------------------------------------------- token scatter
__global__ void scatter_kernel(const int* __restrict__ tk_i,
                               int* __restrict__ meta,
                               int* __restrict__ tok_id,
                               int* __restrict__ slot_of) {
  int t = blockIdx.x * 256 + threadIdx.x;
  if (t >= TOK) return;
  #pragma unroll
  for (int k = 0; k < 2; k++) {
    int e = tk_i[2 * t + k];
    int pos = atomicAdd(&meta[8 + e], 1);
    int slot = meta[16 + e] + pos;
    tok_id[slot] = t;
    slot_of[2 * t + k] = slot;
  }
}

// ---------------------------------------------------- WMMA GEMM template
// C[M,N](f16) = epilogue( A[M,K](f16) * W[N,K](f32->f16)^T + bias[N] )
// Double-buffered LDS, async global->LDS A-tile copies, one barrier/K-step.
template <bool EXPERT, bool GATHER, bool DOGELU>
__global__ __launch_bounds__(256)
void wmma_gemm_kernel(const _Float16* __restrict__ A,
                      const float* __restrict__ W,
                      const float* __restrict__ bias,
                      _Float16* __restrict__ C,
                      int Kdim, int Ndim,
                      const int* __restrict__ tok_id,
                      const int* __restrict__ seg) {
  __shared__ _Float16 As[2][BM * LDA];
  __shared__ _Float16 Bs[2][BN * LDB];
  const int tid = threadIdx.x;
  const int wid = tid >> 5;
  const int lane = tid & 31;
  const int rowBase = blockIdx.x * BM;
  const int n0 = blockIdx.y * BN;

  if (EXPERT) {
    if (rowBase >= seg[8]) return;            // beyond populated segments
    int e = 0;
    #pragma unroll
    for (int i = 0; i < NEXP; i++) if (rowBase >= seg[i]) e = i;
    W    += (size_t)e * Ndim * Kdim;          // e_w1:[E,H,D] / e_w2:[E,O,H]
    bias += (size_t)e * Ndim;
  }

  // Per-thread A-copy slots: 4 chunks of 16B (BM*BK halves / 256 threads);
  // gather indirection resolved once (token ids constant across K).
  int a_r[4], a_o8[4];
  const _Float16* a_src[4];
  #pragma unroll
  for (int it = 0; it < 4; it++) {
    int c = tid + it * 256;
    a_r[it]  = c >> 2;
    a_o8[it] = (c & 3) * 8;
    int row;
    if (GATHER) row = tok_id[rowBase + a_r[it]];        // may be -1 (padding)
    else        row = rowBase + a_r[it];
    a_src[it] = (row >= 0) ? (A + (size_t)row * Kdim + a_o8[it]) : nullptr;
  }
  // Per-thread B slot: 8 consecutive f32 of one weight row.
  const int b_r  = tid >> 2;
  const int b_c8 = (tid & 3) * 8;
  const float* b_src = W + (size_t)(n0 + b_r) * Kdim + b_c8;

  const int ln = lane & 15;
  const int kb = (lane >> 4) * 8;
  v8f acc[8] = {};

  // ---- prologue: stage tile k0 = 0 into buffer 0
  #pragma unroll
  for (int it = 0; it < 4; it++) {
    _Float16* dst = &As[0][a_r[it] * LDA + a_o8[it]];
    if (a_src[it]) async_copy16((unsigned)(uintptr_t)dst, a_src[it]);
    else           *(int4*)dst = make_int4(0, 0, 0, 0);
  }
  {
    float4 f0 = *(const float4*)(b_src);
    float4 f1 = *(const float4*)(b_src + 4);
    __align__(16) _Float16 t8[8] = {
      (_Float16)f0.x, (_Float16)f0.y, (_Float16)f0.z, (_Float16)f0.w,
      (_Float16)f1.x, (_Float16)f1.y, (_Float16)f1.z, (_Float16)f1.w };
    *(int4*)&Bs[0][b_r * LDB + b_c8] = *(const int4*)t8;
  }
  WAIT_ASYNC();
  __syncthreads();

  for (int k0 = 0; k0 < Kdim; k0 += BK) {
    const int  p  = (k0 / BK) & 1;
    const int  np = p ^ 1;
    const bool hasNext = (k0 + BK) < Kdim;

    float4 nf0, nf1;
    if (hasNext) {
      // issue next A tile (async, straight to LDS) + next W loads to regs
      #pragma unroll
      for (int it = 0; it < 4; it++) {
        _Float16* dst = &As[np][a_r[it] * LDA + a_o8[it]];
        if (a_src[it]) async_copy16((unsigned)(uintptr_t)dst, a_src[it] + k0 + BK);
        else           *(int4*)dst = make_int4(0, 0, 0, 0);
      }
      nf0 = *(const float4*)(b_src + k0 + BK);
      nf1 = *(const float4*)(b_src + k0 + BK + 4);
      if (k0 + 2 * BK < Kdim) __builtin_prefetch(b_src + k0 + 2 * BK, 0, 1);
    }

    // compute on buffer p: load all fragments, then 8 back-to-back WMMAs
    const _Float16* ap = &As[p][(wid * 32 + ln) * LDA];
    v16h a0 = ld_frag(ap, kb);
    v16h a1 = ld_frag(ap + 16 * LDA, kb);
    v16h b0 = ld_frag(&Bs[p][(     ln) * LDB], kb);
    v16h b1 = ld_frag(&Bs[p][(16 + ln) * LDB], kb);
    v16h b2 = ld_frag(&Bs[p][(32 + ln) * LDB], kb);
    v16h b3 = ld_frag(&Bs[p][(48 + ln) * LDB], kb);
    acc[0] = __builtin_amdgcn_wmma_f32_16x16x32_f16(false, a0, false, b0, (short)0, acc[0], false, false);
    acc[1] = __builtin_amdgcn_wmma_f32_16x16x32_f16(false, a0, false, b1, (short)0, acc[1], false, false);
    acc[2] = __builtin_amdgcn_wmma_f32_16x16x32_f16(false, a0, false, b2, (short)0, acc[2], false, false);
    acc[3] = __builtin_amdgcn_wmma_f32_16x16x32_f16(false, a0, false, b3, (short)0, acc[3], false, false);
    acc[4] = __builtin_amdgcn_wmma_f32_16x16x32_f16(false, a1, false, b0, (short)0, acc[4], false, false);
    acc[5] = __builtin_amdgcn_wmma_f32_16x16x32_f16(false, a1, false, b1, (short)0, acc[5], false, false);
    acc[6] = __builtin_amdgcn_wmma_f32_16x16x32_f16(false, a1, false, b2, (short)0, acc[6], false, false);
    acc[7] = __builtin_amdgcn_wmma_f32_16x16x32_f16(false, a1, false, b3, (short)0, acc[7], false, false);

    if (hasNext) {
      __align__(16) _Float16 t8[8] = {
        (_Float16)nf0.x, (_Float16)nf0.y, (_Float16)nf0.z, (_Float16)nf0.w,
        (_Float16)nf1.x, (_Float16)nf1.y, (_Float16)nf1.z, (_Float16)nf1.w };
      *(int4*)&Bs[np][b_r * LDB + b_c8] = *(const int4*)t8;
      WAIT_ASYNC();   // A tile for next iter fully resident in LDS
    }
    __syncthreads();
  }

  // C layout: lanes 0-15 -> M=r, N=lane; lanes 16-31 -> M=8+r, N=lane-16
  const int mo = (lane >> 4) * 8;
  #pragma unroll
  for (int sub = 0; sub < 2; sub++) {
    #pragma unroll
    for (int j = 0; j < 4; j++) {
      int col = n0 + j * 16 + ln;
      float bv = bias[col];
      #pragma unroll
      for (int r = 0; r < 8; r++) {
        float v = acc[sub * 4 + j][r] + bv;
        if (DOGELU) v = 0.5f * v * (1.0f + erff(v * 0.70710678118f));  // exact GELU
        int row = rowBase + wid * 32 + sub * 16 + mo + r;
        C[(size_t)row * Ndim + col] = (_Float16)v;
      }
    }
  }
}

// --------------------------------------------- combine top-2 + LN2 out
__global__ __launch_bounds__(256)
void combine_ln2_kernel(const _Float16* __restrict__ shout,
                        const _Float16* __restrict__ eout,
                        const float* __restrict__ tk_w,
                        const int* __restrict__ slot_of,
                        const float* __restrict__ g2,
                        const float* __restrict__ b2,
                        float* __restrict__ out) {
  __shared__ float red[16];
  const int t = blockIdx.x, tid = threadIdx.x;
  const int wid = tid >> 5, lane = tid & 31;
  const size_t s0 = (size_t)slot_of[2 * t];
  const size_t s1 = (size_t)slot_of[2 * t + 1];
  const float w0 = tk_w[2 * t], w1 = tk_w[2 * t + 1];

  float vl[4], sum = 0.f, sq = 0.f;
  #pragma unroll
  for (int i = 0; i < 4; i++) {
    int o = tid + i * 256;
    float v = (float)shout[(size_t)t * OUTD + o] * INV9
            + w0 * (float)eout[s0 * OUTD + o]
            + w1 * (float)eout[s1 * OUTD + o];
    vl[i] = v; sum += v; sq += v * v;
  }
  sum = wave_sum(sum); sq = wave_sum(sq);
  if (lane == 0) { red[wid] = sum; red[8 + wid] = sq; }
  __syncthreads();
  if (tid == 0) {
    float s = 0.f, q = 0.f;
    for (int i = 0; i < 8; i++) { s += red[i]; q += red[8 + i]; }
    red[0] = s; red[8] = q;
  }
  __syncthreads();
  const float mean = red[0] / OUTD;
  const float rstd = rsqrtf(red[8] / OUTD - mean * mean + EPSV);
  #pragma unroll
  for (int i = 0; i < 4; i++) {
    int o = tid + i * 256;
    out[(size_t)t * OUTD + o] = (vl[i] - mean) * rstd * g2[o] + b2[o];
  }
}

// ---------------------------------------------------------------- launch
extern "C" void kernel_launch(void* const* d_in, const int* in_sizes, int n_in,
                              void* d_out, int out_size, void* d_ws, size_t ws_size,
                              hipStream_t stream) {
  const float* x    = (const float*)d_in[0];
  const float* ln1g = (const float*)d_in[1];
  const float* ln1b = (const float*)d_in[2];
  const float* rw   = (const float*)d_in[3];
  const float* rb   = (const float*)d_in[4];
  const float* shw1 = (const float*)d_in[5];
  const float* shb1 = (const float*)d_in[6];
  const float* shw2 = (const float*)d_in[7];
  const float* shb2 = (const float*)d_in[8];
  const float* ew1  = (const float*)d_in[9];
  const float* eb1  = (const float*)d_in[10];
  const float* ew2  = (const float*)d_in[11];
  const float* eb2  = (const float*)d_in[12];
  const float* ln2g = (const float*)d_in[13];
  const float* ln2b = (const float*)d_in[14];
  float* out = (float*)d_out;

  // workspace carve-out (~95 MB)
  char* w = (char*)d_ws;
  _Float16* xh    = (_Float16*)w; w += (size_t)TOK * DIM  * 2;  // x_norm f16
  _Float16* hsh   = (_Float16*)w; w += (size_t)TOK * HID  * 2;  // shared hidden
  _Float16* hexp  = (_Float16*)w; w += (size_t)CAP * HID  * 2;  // expert hidden (per slot)
  _Float16* shout = (_Float16*)w; w += (size_t)TOK * OUTD * 2;  // shared out
  _Float16* eout  = (_Float16*)w; w += (size_t)CAP * OUTD * 2;  // expert out (per slot)
  int*   tk_i     = (int*)w;   w += (size_t)TOK * 2 * 4;
  float* tk_w     = (float*)w; w += (size_t)TOK * 2 * 4;
  int*   tok_id   = (int*)w;   w += (size_t)CAP * 4;
  int*   slot_of  = (int*)w;   w += (size_t)TOK * 2 * 4;
  int*   meta     = (int*)w;   w += 64 * 4;  // counts/cursor/segments

  init_kernel<<<(CAP + 255) / 256, 256, 0, stream>>>(tok_id, meta);
  ln_router_kernel<<<TOK, 256, 0, stream>>>(x, ln1g, ln1b, rw, rb, xh, tk_i, tk_w, meta);
  offsets_kernel<<<1, 32, 0, stream>>>(meta);
  scatter_kernel<<<(TOK + 255) / 256, 256, 0, stream>>>(tk_i, meta, tok_id, slot_of);

  // shared expert MLP
  wmma_gemm_kernel<false, false, true><<<dim3(TOK / BM, HID / BN), 256, 0, stream>>>(
      xh, shw1, shb1, hsh, DIM, HID, nullptr, meta + 16);
  wmma_gemm_kernel<false, false, false><<<dim3(TOK / BM, OUTD / BN), 256, 0, stream>>>(
      hsh, shw2, shb2, shout, HID, OUTD, nullptr, meta + 16);

  // routed experts (grouped, only top-2 per token computed)
  wmma_gemm_kernel<true, true, true><<<dim3(CAP / BM, HID / BN), 256, 0, stream>>>(
      xh, ew1, eb1, hexp, DIM, HID, tok_id, meta + 16);
  wmma_gemm_kernel<true, false, false><<<dim3(CAP / BM, OUTD / BN), 256, 0, stream>>>(
      hexp, ew2, eb2, eout, HID, OUTD, tok_id, meta + 16);

  combine_ln2_kernel<<<TOK, 256, 0, stream>>>(shout, eout, tk_w, slot_of, ln2g, ln2b, out);
}
